// BSRBF_KANLayer_50508815401098
// MI455X (gfx1250) — matrix operand order repaired and checked by reading, not compiled, not run on hardware
//
#include <hip/hip_runtime.h>

typedef __attribute__((ext_vector_type(16))) __bf16 v16bf;
typedef __attribute__((ext_vector_type(8)))  __bf16 v8bf;
typedef __attribute__((ext_vector_type(4)))  __bf16 v4bf;
typedef __attribute__((ext_vector_type(8)))  float  v8f;

#define D_IN   512
#define NB     8
#define K_TOT  4608          // 512 base + 4096 spline
#define N_OUT  512
#define M_TOT  16384
#define TM     64            // rows per workgroup
#define TK     64            // K chunk
#define LDA    72            // LDS stride (elements) for A tile
#define LDB    72            // LDS stride (elements) for B tile
#define LN_EPS 1e-5f

// ---------------- kernel 1: per-row LayerNorm stats -------------------------
__global__ __launch_bounds__(256) void kan_ln_stats(const float* __restrict__ x,
                                                    float2* __restrict__ stats) {
    __shared__ float ssum[256];
    __shared__ float ssq[256];
    const int t = threadIdx.x;
    const int row = blockIdx.x;
    const float* xr = x + (size_t)row * D_IN;
    float a = xr[t], b = xr[t + 256];
    ssum[t] = a + b;
    ssq[t]  = a * a + b * b;
    __syncthreads();
    for (int off = 128; off > 0; off >>= 1) {
        if (t < off) { ssum[t] += ssum[t + off]; ssq[t] += ssq[t + off]; }
        __syncthreads();
    }
    if (t == 0) {
        float mu  = ssum[0] * (1.0f / D_IN);
        float var = ssq[0] * (1.0f / D_IN) - mu * mu;
        float2 st; st.x = mu; st.y = rsqrtf(var + LN_EPS);
        stats[row] = st;
    }
}

// -------- kernel 2: one-shot weight convert: Wb[n][k] bf16, K = 512|4096 ----
__global__ __launch_bounds__(256) void kan_wconv(const float* __restrict__ bw,
                                                 const float* __restrict__ sw,
                                                 __bf16* __restrict__ wb) {
    const int g  = blockIdx.x * 256 + threadIdx.x;   // 16 elements per thread
    const int e0 = g * 16;
    const int n  = e0 / K_TOT;
    const int k  = e0 - n * K_TOT;                   // multiple of 16, never straddles 512
    const float* src = (k < D_IN) ? (bw + (size_t)n * D_IN + k)
                                  : (sw + (size_t)n * 4096 + (k - D_IN));
    __bf16* dst = wb + (size_t)n * K_TOT + k;
#pragma unroll
    for (int i = 0; i < 4; ++i) {
        float4 v = ((const float4*)src)[i];
        v4bf p;
        p[0] = (__bf16)v.x; p[1] = (__bf16)v.y;
        p[2] = (__bf16)v.z; p[3] = (__bf16)v.w;
        *(v4bf*)(dst + i * 4) = p;
    }
}

// ------------- basis: order-3 B-spline (uniform knots) + RBF ----------------
__device__ __forceinline__ void bsrbf8(float xn, float* __restrict__ bs) {
    float b0[11];
#pragma unroll
    for (int i = 0; i < 11; ++i) {
        float gl = -3.3f + 0.6f * (float)i;
        b0[i] = (xn >= gl && xn < gl + 0.6f) ? 1.0f : 0.0f;
    }
    const float inv1 = 1.0f / 0.6f, inv2 = 1.0f / 1.2f, inv3 = 1.0f / 1.8f;
    float b1[10];
#pragma unroll
    for (int i = 0; i < 10; ++i) {
        float gi = -3.3f + 0.6f * (float)i;
        b1[i] = (xn - gi) * inv1 * b0[i] + ((gi + 1.2f) - xn) * inv1 * b0[i + 1];
    }
    float b2[9];
#pragma unroll
    for (int i = 0; i < 9; ++i) {
        float gi = -3.3f + 0.6f * (float)i;
        b2[i] = (xn - gi) * inv2 * b1[i] + ((gi + 1.8f) - xn) * inv2 * b1[i + 1];
    }
    float b3[8];
#pragma unroll
    for (int i = 0; i < 8; ++i) {
        float gi = -3.3f + 0.6f * (float)i;
        b3[i] = (xn - gi) * inv3 * b2[i] + ((gi + 2.4f) - xn) * inv3 * b2[i + 1];
    }
    const float invd = 7.0f / 3.0f;
#pragma unroll
    for (int j = 0; j < 8; ++j) {
        float c = -1.5f + (3.0f / 7.0f) * (float)j;
        float tt = (xn - c) * invd;
        bs[j] = b3[j] + __expf(-tt * tt);
    }
}

// -------- kernel 3: fused basis-gen + GEMM (full N per WG) via WMMA ---------
__global__ __launch_bounds__(256) void kan_main(const float* __restrict__ x,
                                                const float* __restrict__ lnw,
                                                const float* __restrict__ lnb,
                                                const __bf16* __restrict__ wb,   // [512][4608] bf16
                                                const float2* __restrict__ stats,
                                                float* __restrict__ out) {
    __shared__ __align__(16) __bf16 As[TM * LDA];       //  9.2 KB
    __shared__ __align__(16) __bf16 Bs[N_OUT * LDB];    // 73.7 KB
    __shared__ float2 st[TM];

    const int t = threadIdx.x;
    const int mbase = blockIdx.x * TM;
    if (t < TM) st[t] = stats[mbase + t];
    __syncthreads();

    const int wave = t >> 5, lane = t & 31;
    const int wm = wave & 1;          // 0..1 : 32-row slab
    const int wn = wave >> 1;         // 0..3 : 128-col slab
    const int lm = lane & 15, lh = lane >> 4;

    v8f acc[2][8];
#pragma unroll
    for (int mi = 0; mi < 2; ++mi)
#pragma unroll
        for (int ni = 0; ni < 8; ++ni)
#pragma unroll
            for (int r = 0; r < 8; ++r) acc[mi][ni][r] = 0.0f;

    for (int kt = 0; kt < K_TOT / TK; ++kt) {
        const int k0 = kt * TK;

        // ---- B tile: async copy bf16 weights -> LDS (2 rows/thread, 128 B each)
        {
#pragma unroll
            for (int rr = 0; rr < 2; ++rr) {
                const int row = t + rr * 256;
                unsigned long long ga =
                    (unsigned long long)(uintptr_t)(wb + (size_t)row * K_TOT + k0);
                unsigned int la =
                    (unsigned int)(uintptr_t)(&Bs[row * LDB]);
#pragma unroll
                for (int i = 0; i < 8; ++i)
                    asm volatile("global_load_async_to_lds_b128 %0, %1, off offset:%2"
                                 :: "v"(la), "v"(ga), "i"(i * 16) : "memory");
            }
        }

        // ---- A tile: generate activations / basis features
        if (k0 < D_IN) {
            // base branch region: relu(layernorm(x))
            const int r = t >> 2;
            const int c0 = (t & 3) * 16;
            const float2 ms = st[r];
            const float* xp = x + (size_t)(mbase + r) * D_IN + k0 + c0;
            const float* wp = lnw + k0 + c0;
            const float* bp = lnb + k0 + c0;
            __bf16* dst = &As[r * LDA + c0];
#pragma unroll
            for (int i = 0; i < 4; ++i) {
                float4 v = ((const float4*)xp)[i];
                float4 w = ((const float4*)wp)[i];
                float4 b = ((const float4*)bp)[i];
                v4bf p;
                p[0] = (__bf16)fmaxf((v.x - ms.x) * ms.y * w.x + b.x, 0.0f);
                p[1] = (__bf16)fmaxf((v.y - ms.x) * ms.y * w.y + b.y, 0.0f);
                p[2] = (__bf16)fmaxf((v.z - ms.x) * ms.y * w.z + b.z, 0.0f);
                p[3] = (__bf16)fmaxf((v.w - ms.x) * ms.y * w.w + b.w, 0.0f);
                *(v4bf*)(dst + i * 4) = p;
            }
        } else {
            // spline branch region: 8 features per (row, d); chunk covers 8 d's
            const int d0 = (k0 - D_IN) >> 3;
#pragma unroll
            for (int it = 0; it < 2; ++it) {
                const int task = t + it * 256;
                const int r = task >> 3;
                const int dl = task & 7;
                const int d = d0 + dl;
                const float2 ms = st[r];
                float xv = x[(size_t)(mbase + r) * D_IN + d];
                float xn = (xv - ms.x) * ms.y * lnw[d] + lnb[d];
                float bs[8];
                bsrbf8(xn, bs);
                v8bf p;
#pragma unroll
                for (int j = 0; j < 8; ++j) p[j] = (__bf16)bs[j];
                *(v8bf*)&As[r * LDA + dl * 8] = p;
            }
        }
        asm volatile("s_wait_asynccnt 0x0" ::: "memory");
        __syncthreads();

        // ---- 2 WMMA K-steps of 32 over the 64-wide chunk
#pragma unroll
        for (int ks = 0; ks < 2; ++ks) {
            const int kk = ks * 32;
            v16bf afrag[2];
#pragma unroll
            for (int mi = 0; mi < 2; ++mi) {
                const __bf16* pa = &As[(wm * 32 + mi * 16 + lm) * LDA + kk + lh * 8];
                v8bf lo = *(const v8bf*)pa;
                v8bf hi = *(const v8bf*)(pa + 16);
                afrag[mi] = __builtin_shufflevector(lo, hi, 0, 1, 2, 3, 4, 5, 6, 7,
                                                    8, 9, 10, 11, 12, 13, 14, 15);
            }
#pragma unroll
            for (int ni = 0; ni < 8; ++ni) {
                const __bf16* pb = &Bs[(wn * 128 + ni * 16 + lm) * LDB + kk + lh * 8];
                v8bf lo = *(const v8bf*)pb;
                v8bf hi = *(const v8bf*)(pb + 16);
                v16bf bfrag = __builtin_shufflevector(lo, hi, 0, 1, 2, 3, 4, 5, 6, 7,
                                                      8, 9, 10, 11, 12, 13, 14, 15);
#pragma unroll
                for (int mi = 0; mi < 2; ++mi)
                    acc[mi][ni] = __builtin_amdgcn_wmma_f32_16x16x32_bf16(
                        false, afrag[mi], false, bfrag,
                        (short)0, acc[mi][ni], false, false);
            }
        }
        __syncthreads();
    }

    // ---- store C: VGPR r -> (M = r or 8+r), N = lane
#pragma unroll
    for (int mi = 0; mi < 2; ++mi)
#pragma unroll
        for (int ni = 0; ni < 8; ++ni) {
            const int mrow = mbase + wm * 32 + mi * 16 + lh * 8;
            const int ncol = wn * 128 + ni * 16 + lm;
#pragma unroll
            for (int r = 0; r < 8; ++r)
                out[(size_t)(mrow + r) * N_OUT + ncol] = acc[mi][ni][r];
        }
}

extern "C" void kernel_launch(void* const* d_in, const int* in_sizes, int n_in,
                              void* d_out, int out_size, void* d_ws, size_t ws_size,
                              hipStream_t stream) {
    const float* x   = (const float*)d_in[0];
    const float* lnw = (const float*)d_in[1];
    const float* lnb = (const float*)d_in[2];
    const float* bw  = (const float*)d_in[3];
    const float* sw  = (const float*)d_in[4];
    float* out = (float*)d_out;

    float2* stats = (float2*)d_ws;                            // 128 KB
    __bf16* wbf   = (__bf16*)((char*)d_ws + 131072);          // 4.5 MB bf16 weights

    kan_ln_stats<<<M_TOT, 256, 0, stream>>>(x, stats);
    kan_wconv<<<(N_OUT * K_TOT) / (256 * 16), 256, 0, stream>>>(bw, sw, wbf);
    kan_main<<<M_TOT / TM, 256, 0, stream>>>(x, lnw, lnb, wbf, stats, out);
}
